// StructureModule_66984309948715
// MI455X (gfx1250) — matrix operand order repaired and checked by reading, not compiled, not run on hardware
//
#include <hip/hip_runtime.h>

// ---------------------------------------------------------------------------
// AlphaFold-style structure module for MI455X (gfx1250, wave32, WMMA).
// All GEMM-like einsums run on v_wmma_f32_16x16x32_bf16 (fp32 accumulate).
// Working set ~60MB fits in 192MB L2 -> latency bound; operand gathers are
// branch-free (clamp+select) and A-side uses b128 vector loads.
// ---------------------------------------------------------------------------

namespace {
constexpr int Lc = 384;   // residues
constexpr int SD = 128;   // single dim
constexpr int ZD = 64;    // pair dim
constexpr int Hn = 16;    // heads
constexpr int Cn = 16;    // head dim
constexpr int Qn = 4;     // query points
constexpr int Pn = 6;     // value points
constexpr int NL = 4;     // layers
constexpr float C_SQU    = 0.25f;                  // 1/sqrt(C)
constexpr float GG_SCALE = 0.11785113019775793f;   // 0.5 * sqrt(2/(9*Q))
}

typedef __attribute__((ext_vector_type(16))) __bf16 v16bf;
typedef __attribute__((ext_vector_type(8)))  float  v8f;

__device__ __forceinline__ v8f wmma_bf16(v16bf a, v16bf b, v8f c) {
  return __builtin_amdgcn_wmma_f32_16x16x32_bf16(false, a, false, b,
                                                 (short)0, c, false, false);
}

__device__ __forceinline__ void pack4(v16bf& v, int o, float4 f) {
  v[o + 0] = (__bf16)f.x; v[o + 1] = (__bf16)f.y;
  v[o + 2] = (__bf16)f.z; v[o + 3] = (__bf16)f.w;
}

// 16-bit A-matrix 16x32: lane = row M; e -> k = e + 16*half (contiguous,
// matching B). Contiguous chunk -> b128 loads.
__device__ __forceinline__ v16bf loadA_tile(const float* rowbase, int half) {
  const float* p = rowbase + (half << 4);
  float4 x0 = *(const float4*)(p);
  float4 x1 = *(const float4*)(p + 4);
  float4 y0 = *(const float4*)(p + 8);
  float4 y1 = *(const float4*)(p + 12);
  v16bf a;
  pack4(a, 0, x0); pack4(a, 4, x1); pack4(a, 8, y0); pack4(a, 12, y1);
  return a;
}

// 16-bit B-matrix 32x16: lane = col N; e -> k = e + 16*half (contiguous, matching A).
__device__ __forceinline__ v16bf loadB_contig(const float* colbase, int half) {
  const float* p = colbase + (half << 4);
  float4 x0 = *(const float4*)(p);
  float4 x1 = *(const float4*)(p + 4);
  float4 x2 = *(const float4*)(p + 8);
  float4 x3 = *(const float4*)(p + 12);
  v16bf b;
  pack4(b, 0, x0); pack4(b, 4, x1); pack4(b, 8, x2); pack4(b, 12, x3);
  return b;
}

__device__ __forceinline__ int kmapB(int e, int half) { return e + (half << 4); }
__device__ __forceinline__ int waveId() {
  return (int)(blockIdx.x * (blockDim.x >> 5) + (threadIdx.x >> 5));
}

// ---------------- LayerNorm: one wave32 per row -----------------------------
__global__ void k_ln_wave(const float* __restrict__ x, float* __restrict__ y,
                          const float* __restrict__ g, const float* __restrict__ b,
                          int rows, int dim) {
  int wv = waveId();
  if (wv >= rows) return;
  int lane = threadIdx.x & 31;
  const float* xr = x + (size_t)wv * dim;
  float s = 0.f, s2 = 0.f;
  for (int i = lane; i < dim; i += 32) { float t = xr[i]; s += t; s2 += t * t; }
#pragma unroll
  for (int o = 16; o > 0; o >>= 1) {
    s  += __shfl_xor(s,  o, 32);
    s2 += __shfl_xor(s2, o, 32);
  }
  float m = s / dim;
  float v = s2 / dim - m * m;
  float inv = rsqrtf(v + 1e-5f);
  float* yr = y + (size_t)wv * dim;
  for (int i = lane; i < dim; i += 32) yr[i] = (xr[i] - m) * inv * g[i] + b[i];
}

// ---------------- Generic GEMM: D[M,N] = A[M,K] @ B[K,N] (+bias)(+D)(relu) --
// M % 16 == 0, K % 32 == 0. One wave per 16x16 tile.
__global__ void k_gemm(const float* __restrict__ A, int lda,
                       const float* __restrict__ B, int ldb,
                       float* __restrict__ D, int ldd,
                       int M, int N, int K,
                       const float* __restrict__ bias, int flags) {
  int nTm = M >> 4, nTn = (N + 15) >> 4;
  int wv = waveId();
  if (wv >= nTm * nTn) return;
  int tm = wv % nTm, tn = wv / nTm;
  int m0 = tm << 4, n0 = tn << 4;
  int lane = threadIdx.x & 31, row = lane & 15, half = lane >> 4;
  int n = n0 + row;
  int nc = (n < N) ? n : (N - 1);     // clamped in-bounds column
  bool nok = (n < N);
  const float* Arow = A + (size_t)(m0 + row) * lda;
  v8f acc = {};
  for (int k0 = 0; k0 < K; k0 += 32) {
    v16bf a = loadA_tile(Arow + k0, half);
    v16bf b;
#pragma unroll
    for (int e = 0; e < 16; ++e) {
      float bv = B[(size_t)(k0 + kmapB(e, half)) * ldb + nc];  // unconditional
      b[e] = (__bf16)(nok ? bv : 0.f);                          // v_cndmask
    }
    acc = wmma_bf16(a, b, acc);
  }
  if (nok) {
#pragma unroll
    for (int r = 0; r < 8; ++r) {
      int m = m0 + r + (half << 3);
      float v = acc[r];
      if (bias) v += bias[n];
      if (flags & 2) v += D[(size_t)m * ldd + n];
      if (flags & 1) v = fmaxf(v, 0.f);
      D[(size_t)m * ldd + n] = v;
    }
  }
}

// ---------------- bias_z = zN[L*L,ZD] @ Wz[ZD,H], scattered to [i][h][j] ----
__global__ void k_biasz(const float* __restrict__ zN, const float* __restrict__ Wz,
                        float* __restrict__ bz) {
  int wv = waveId();
  if (wv >= (Lc * Lc) / 16) return;
  int m0 = wv << 4;
  int lane = threadIdx.x & 31, row = lane & 15, half = lane >> 4;
  const float* Arow = zN + (size_t)(m0 + row) * ZD;
  v8f acc = {};
  for (int k0 = 0; k0 < ZD; k0 += 32) {
    v16bf a = loadA_tile(Arow + k0, half);
    v16bf b;
#pragma unroll
    for (int e = 0; e < 16; ++e)
      b[e] = (__bf16)Wz[(size_t)(k0 + kmapB(e, half)) * Hn + row];
    acc = wmma_bf16(a, b, acc);
  }
  int h = row;
#pragma unroll
  for (int r = 0; r < 8; ++r) {
    int m = m0 + r + (half << 3);
    int i = m / Lc, j = m % Lc;
    bz[((size_t)i * Hn + h) * Lc + j] = acc[r];
  }
}

// ---------------- rotate/translate points; build padded K=32 att operands ---
// qext[i,h,0:16] = q1*C_SQU ; [16:28] = (rot@q2+t)*2g ; [28:32]=0
// kext[j,h,0:16] = k1       ; [16:28] = (rot@k2+t)    ; [28:32]=0
__global__ void k_points(const float* __restrict__ q1, const float* __restrict__ k1,
                         const float* __restrict__ q2, const float* __restrict__ k2,
                         const float* __restrict__ v3, const float* __restrict__ rot,
                         const float* __restrict__ trn, const float* __restrict__ gamma,
                         float* __restrict__ qext, float* __restrict__ kext,
                         float* __restrict__ qn, float* __restrict__ kn,
                         float* __restrict__ v3t) {
  int t = blockIdx.x * blockDim.x + threadIdx.x;
  if (t >= Lc * Hn) return;
  int i = t / Hn, h = t % Hn;
  float ga = gamma[h];
  float sp = (ga > 20.f) ? ga : log1pf(expf(ga));   // softplus
  float gg = sp * GG_SCALE;
  float R[9], T[3];
#pragma unroll
  for (int a = 0; a < 9; ++a) R[a] = rot[i * 9 + a];
#pragma unroll
  for (int a = 0; a < 3; ++a) T[a] = trn[i * 3 + a];

  float* qe = qext + (size_t)t * 32;
  float* ke = kext + (size_t)t * 32;
#pragma unroll
  for (int c = 0; c < 16; ++c) {
    qe[c] = q1[(size_t)i * (Hn * Cn) + h * Cn + c] * C_SQU;
    ke[c] = k1[(size_t)i * (Hn * Cn) + h * Cn + c];
  }
  float sq = 0.f, sk = 0.f;
  for (int q = 0; q < Qn; ++q) {
    const float* pq = q2 + (size_t)i * (Hn * Qn * 3) + h * (Qn * 3) + q * 3;
    const float* pk = k2 + (size_t)i * (Hn * Qn * 3) + h * (Qn * 3) + q * 3;
#pragma unroll
    for (int d = 0; d < 3; ++d) {
      float oq = pq[0] * R[0 * 3 + d] + pq[1] * R[1 * 3 + d] + pq[2] * R[2 * 3 + d] + T[d];
      float ok = pk[0] * R[0 * 3 + d] + pk[1] * R[1 * 3 + d] + pk[2] * R[2 * 3 + d] + T[d];
      qe[16 + q * 3 + d] = oq * (2.f * gg);
      ke[16 + q * 3 + d] = ok;
      sq += oq * oq;
      sk += ok * ok;
    }
  }
#pragma unroll
  for (int c = 28; c < 32; ++c) { qe[c] = 0.f; ke[c] = 0.f; }
  qn[t] = sq;
  kn[t] = sk;
  for (int p = 0; p < Pn; ++p) {
    const float* pv = v3 + (size_t)i * (Hn * Pn * 3) + h * (Pn * 3) + p * 3;
#pragma unroll
    for (int d = 0; d < 3; ++d) {
      v3t[((size_t)i * Hn + h) * 18 + p * 3 + d] =
          pv[0] * R[0 * 3 + d] + pv[1] * R[1 * 3 + d] + pv[2] * R[2 * 3 + d] + T[d];
    }
  }
}

// ---------------- attention logits: one WMMA per 16x16 tile per head --------
// att[i,j,h] = qext[i,h,:].kext[j,h,:] + bias_z[i,j,h] - g*(qn[i,h]+kn[j,h])
__global__ void k_att(const float* __restrict__ qext, const float* __restrict__ kext,
                      const float* __restrict__ qn, const float* __restrict__ kn,
                      const float* __restrict__ bz, const float* __restrict__ gamma,
                      float* __restrict__ att) {
  constexpr int TI = Lc / 16, TJ = Lc / 16;
  int wv = waveId();
  if (wv >= TI * TJ * Hn) return;
  int h = wv / (TI * TJ);
  int rem = wv % (TI * TJ);
  int ti = rem / TJ, tj = rem % TJ;
  int m0 = ti << 4, n0 = tj << 4;
  int lane = threadIdx.x & 31, row = lane & 15, half = lane >> 4;
  float ga = gamma[h];
  float sp = (ga > 20.f) ? ga : log1pf(expf(ga));
  float gg = sp * GG_SCALE;
  v8f acc;
  int j = n0 + row;
#pragma unroll
  for (int r = 0; r < 8; ++r) {
    int i = m0 + r + (half << 3);
    acc[r] = bz[((size_t)i * Hn + h) * Lc + j] - gg * (qn[i * Hn + h] + kn[j * Hn + h]);
  }
  v16bf a = loadA_tile(qext + ((size_t)(m0 + row) * Hn + h) * 32, half);
  v16bf b = loadB_contig(kext + ((size_t)(n0 + row) * Hn + h) * 32, half);
  acc = wmma_bf16(a, b, acc);
#pragma unroll
  for (int r = 0; r < 8; ++r) {
    int i = m0 + r + (half << 3);
    att[((size_t)i * Hn + h) * Lc + j] = acc[r];
  }
}

// ---------------- o1[i,h,d] = sum_j att[i,j,h] * zN[i,j,d] ------------------
__global__ void k_o1(const float* __restrict__ att, const float* __restrict__ zN,
                     float* __restrict__ o1) {
  int wv = waveId();
  if (wv >= Lc * (ZD / 16)) return;
  int i = wv >> 2, tn = wv & 3, n0 = tn << 4;
  int lane = threadIdx.x & 31, row = lane & 15, half = lane >> 4;
  const float* Arow = att + ((size_t)i * Hn + row) * Lc;
  const float* Bcol = zN + (size_t)i * Lc * ZD + n0 + row;
  v8f acc = {};
  for (int k0 = 0; k0 < Lc; k0 += 32) {
    v16bf a = loadA_tile(Arow + k0, half);
    v16bf b;
#pragma unroll
    for (int e = 0; e < 16; ++e)
      b[e] = (__bf16)Bcol[(size_t)(k0 + kmapB(e, half)) * ZD];
    acc = wmma_bf16(a, b, acc);
  }
#pragma unroll
  for (int r = 0; r < 8; ++r) {
    int hh = r + (half << 3);
    o1[(size_t)i * (Hn * ZD) + hh * ZD + n0 + row] = acc[r];
  }
}

// ---------------- o2[d,h,c] = sum_a att[d,a,h] * v1[a,h,c] ------------------
__global__ void k_o2(const float* __restrict__ att, const float* __restrict__ v1,
                     float* __restrict__ o2) {
  constexpr int TM = Lc / 16;
  int wv = waveId();
  if (wv >= TM * Hn) return;
  int h = wv / TM, tm = wv % TM, m0 = tm << 4;
  int lane = threadIdx.x & 31, row = lane & 15, half = lane >> 4;
  const float* Arow = att + ((size_t)(m0 + row) * Hn + h) * Lc;
  const float* Bcol = v1 + h * Cn + row;
  v8f acc = {};
  for (int k0 = 0; k0 < Lc; k0 += 32) {
    v16bf a = loadA_tile(Arow + k0, half);
    v16bf b;
#pragma unroll
    for (int e = 0; e < 16; ++e)
      b[e] = (__bf16)Bcol[(size_t)(k0 + kmapB(e, half)) * (Hn * Cn)];
    acc = wmma_bf16(a, b, acc);
  }
#pragma unroll
  for (int r = 0; r < 8; ++r) {
    int m = m0 + r + (half << 3);
    o2[(size_t)m * (Hn * Cn) + h * Cn + row] = acc[r];
  }
}

// ---------------- tmp[g,h,pt] = sum_v att[g,v,h] * v3t[v,h,pt] --------------
__global__ void k_tmp(const float* __restrict__ att, const float* __restrict__ v3t,
                      float* __restrict__ tb) {
  constexpr int TM = Lc / 16;
  int wv = waveId();
  if (wv >= TM * 2 * Hn) return;
  int h = wv / (TM * 2);
  int rem = wv % (TM * 2);
  int tm = rem >> 1, tn = rem & 1;
  int m0 = tm << 4, n0 = tn << 4;
  int lane = threadIdx.x & 31, row = lane & 15, half = lane >> 4;
  int n = n0 + row;
  int nc = (n < 18) ? n : 17;
  bool nok = (n < 18);
  const float* Arow = att + ((size_t)(m0 + row) * Hn + h) * Lc;
  const float* Bcol = v3t + h * 18 + nc;
  v8f acc = {};
  for (int k0 = 0; k0 < Lc; k0 += 32) {
    v16bf a = loadA_tile(Arow + k0, half);
    v16bf b;
#pragma unroll
    for (int e = 0; e < 16; ++e) {
      float bv = Bcol[(size_t)(k0 + kmapB(e, half)) * (Hn * 18)];
      b[e] = (__bf16)(nok ? bv : 0.f);
    }
    acc = wmma_bf16(a, b, acc);
  }
  if (nok) {
#pragma unroll
    for (int r = 0; r < 8; ++r) {
      int m = m0 + r + (half << 3);
      tb[((size_t)m * Hn + h) * 18 + n] = acc[r];
    }
  }
}

// ---------------- o3 = rot^T applied to (tmp - trans) -----------------------
__global__ void k_o3(const float* __restrict__ tb, const float* __restrict__ rot,
                     const float* __restrict__ trn, float* __restrict__ o3) {
  int t = blockIdx.x * blockDim.x + threadIdx.x;
  if (t >= Lc * Hn) return;
  int g = t / Hn, h = t % Hn;
  float R[9], T[3];
#pragma unroll
  for (int a = 0; a < 9; ++a) R[a] = rot[g * 9 + a];
#pragma unroll
  for (int a = 0; a < 3; ++a) T[a] = trn[g * 3 + a];
  for (int p = 0; p < Pn; ++p) {
    float w0 = tb[((size_t)g * Hn + h) * 18 + p * 3 + 0] - T[0];
    float w1 = tb[((size_t)g * Hn + h) * 18 + p * 3 + 1] - T[1];
    float w2 = tb[((size_t)g * Hn + h) * 18 + p * 3 + 2] - T[2];
#pragma unroll
    for (int d = 0; d < 3; ++d) {  // o3[d] = sum_a w[a] * rot[g,d,a]
      o3[(size_t)g * (Hn * Pn * 3) + h * (Pn * 3) + p * 3 + d] =
          w0 * R[d * 3 + 0] + w1 * R[d * 3 + 1] + w2 * R[d * 3 + 2];
    }
  }
}

// ---------------- frame init & quaternion update ----------------------------
__global__ void k_frame_init(float* __restrict__ rot, float* __restrict__ trn) {
  int i = blockIdx.x * blockDim.x + threadIdx.x;
  if (i >= Lc) return;
#pragma unroll
  for (int a = 0; a < 9; ++a) rot[i * 9 + a] = (a % 4 == 0) ? 1.f : 0.f;
#pragma unroll
  for (int a = 0; a < 3; ++a) trn[i * 3 + a] = 0.f;
}

__global__ void k_frame_update(const float* __restrict__ pred,
                               float* __restrict__ rot, float* __restrict__ trn) {
  int i = blockIdx.x * blockDim.x + threadIdx.x;
  if (i >= Lc) return;
  float x = pred[i * 6 + 0], y = pred[i * 6 + 1], z = pred[i * 6 + 2];
  float inv = 1.f / (1.f + x * x + y * y + z * z);
  float n[9] = {
      (1.f + x * x - y * y - z * z), 2.f * (x * y - z),             2.f * (x * z + y),
      2.f * (x * y + z),             (1.f - x * x + y * y - z * z), 2.f * (y * z - x),
      2.f * (x * z - y),             2.f * (y * z + x),             (1.f - x * x - y * y + z * z)};
  float R[9];
#pragma unroll
  for (int a = 0; a < 9; ++a) R[a] = rot[i * 9 + a];
  float Rn[9];
#pragma unroll
  for (int j = 0; j < 3; ++j)
#pragma unroll
    for (int d = 0; d < 3; ++d) {
      float acc = 0.f;
#pragma unroll
      for (int a = 0; a < 3; ++a) acc += n[j * 3 + a] * inv * R[a * 3 + d];
      Rn[j * 3 + d] = acc;
    }
  float tv[3] = {pred[i * 6 + 3], pred[i * 6 + 4], pred[i * 6 + 5]};
#pragma unroll
  for (int d = 0; d < 3; ++d) {
    float acc = trn[i * 3 + d];
#pragma unroll
    for (int a = 0; a < 3; ++a) acc += tv[a] * R[a * 3 + d];
    trn[i * 3 + d] = acc;
  }
#pragma unroll
  for (int a = 0; a < 9; ++a) rot[i * 9 + a] = Rn[a];
}

// ---------------------------------------------------------------------------
static inline int cdiv(int a, int b) { return (a + b - 1) / b; }

static void gemm(hipStream_t st, const float* A, int lda, const float* B, int ldb,
                 float* D, int ldd, int M, int N, int K, const float* bias, int flags) {
  int tiles = (M >> 4) * ((N + 15) >> 4);
  k_gemm<<<cdiv(tiles, 8), 256, 0, st>>>(A, lda, B, ldb, D, ldd, M, N, K, bias, flags);
}

extern "C" void kernel_launch(void* const* d_in, const int* in_sizes, int n_in,
                              void* d_out, int out_size, void* d_ws, size_t ws_size,
                              hipStream_t stream) {
  (void)in_sizes; (void)n_in; (void)out_size; (void)ws_size;
  const float* s_in = (const float*)d_in[0];
  const float* z_in = (const float*)d_in[1];
  // params in setup_inputs dict order
  const float* ln_s_g = (const float*)d_in[2];
  const float* ln_s_b = (const float*)d_in[3];
  const float* ln_z_g = (const float*)d_in[4];
  const float* ln_z_b = (const float*)d_in[5];
  const float* Wq1 = (const float*)d_in[6];
  const float* Wk1 = (const float*)d_in[7];
  const float* Wv1 = (const float*)d_in[8];
  const float* Wq2 = (const float*)d_in[9];
  const float* Wk2 = (const float*)d_in[10];
  const float* Wv3 = (const float*)d_in[11];
  const float* Wz  = (const float*)d_in[12];
  const float* Wl1 = (const float*)d_in[13];
  const float* bl1 = (const float*)d_in[14];
  const float* Wl2 = (const float*)d_in[15];
  const float* bl2 = (const float*)d_in[16];
  const float* Wl3 = (const float*)d_in[17];
  const float* bl3 = (const float*)d_in[18];
  const float* gamma = (const float*)d_in[19];
  const float* tn1g = (const float*)d_in[20];
  const float* tn1b = (const float*)d_in[21];
  const float* Wt1 = (const float*)d_in[22];
  const float* bt1 = (const float*)d_in[23];
  const float* Wt2 = (const float*)d_in[24];
  const float* bt2 = (const float*)d_in[25];
  const float* Wt3 = (const float*)d_in[26];
  const float* bt3 = (const float*)d_in[27];
  const float* tn2g = (const float*)d_in[28];
  const float* tn2b = (const float*)d_in[29];
  const float* Wbb = (const float*)d_in[30];
  const float* bbb = (const float*)d_in[31];

  float* s = (float*)d_out;  // s lives in d_out (L x SD)

  // workspace carve-up (fp32); all offsets 16B aligned
  float* w = (float*)d_ws;
  float* zN  = w; w += (size_t)Lc * Lc * ZD;         // 9.4M floats
  float* bz  = w; w += (size_t)Lc * Hn * Lc;         // [i][h][j]
  float* att = w; w += (size_t)Lc * Hn * Lc;         // [i][h][j]
  float* q1  = w; w += (size_t)Lc * Hn * Cn;
  float* k1  = w; w += (size_t)Lc * Hn * Cn;
  float* v1  = w; w += (size_t)Lc * Hn * Cn;
  float* q2  = w; w += (size_t)Lc * Hn * Qn * 3;
  float* k2  = w; w += (size_t)Lc * Hn * Qn * 3;
  float* v3  = w; w += (size_t)Lc * Hn * Pn * 3;
  float* qext = w; w += (size_t)Lc * Hn * 32;        // padded att A-operand
  float* kext = w; w += (size_t)Lc * Hn * 32;        // padded att B-operand
  float* qn  = w; w += (size_t)Lc * Hn;
  float* kn  = w; w += (size_t)Lc * Hn;
  float* v3t = w; w += (size_t)Lc * Hn * 18;
  float* o1  = w; w += (size_t)Lc * Hn * ZD;
  float* o2  = w; w += (size_t)Lc * Hn * Cn;
  float* tb  = w; w += (size_t)Lc * Hn * 18;
  float* o3  = w; w += (size_t)Lc * Hn * Pn * 3;
  float* t0  = w; w += (size_t)Lc * SD;
  float* t1  = w; w += (size_t)Lc * SD;
  float* t2  = w; w += (size_t)Lc * SD;
  float* pred = w; w += (size_t)Lc * 8;
  float* rot = w; w += (size_t)Lc * 9;
  float* trn = w; w += (size_t)Lc * 3;

  // ---- pre-loop ----
  k_frame_init<<<cdiv(Lc, 256), 256, 0, stream>>>(rot, trn);
  k_ln_wave<<<cdiv(Lc, 8), 256, 0, stream>>>(s_in, s, ln_s_g, ln_s_b, Lc, SD);
  k_ln_wave<<<cdiv(Lc * Lc, 8), 256, 0, stream>>>(z_in, zN, ln_z_g, ln_z_b, Lc * Lc, ZD);
  k_biasz<<<cdiv((Lc * Lc) / 16, 8), 256, 0, stream>>>(zN, Wz, bz);

  for (int layer = 0; layer < NL; ++layer) {
    // projections from current s
    gemm(stream, s, SD, Wq1, Hn * Cn, q1, Hn * Cn, Lc, Hn * Cn, SD, nullptr, 0);
    gemm(stream, s, SD, Wk1, Hn * Cn, k1, Hn * Cn, Lc, Hn * Cn, SD, nullptr, 0);
    gemm(stream, s, SD, Wv1, Hn * Cn, v1, Hn * Cn, Lc, Hn * Cn, SD, nullptr, 0);
    gemm(stream, s, SD, Wq2, Hn * Qn * 3, q2, Hn * Qn * 3, Lc, Hn * Qn * 3, SD, nullptr, 0);
    gemm(stream, s, SD, Wk2, Hn * Qn * 3, k2, Hn * Qn * 3, Lc, Hn * Qn * 3, SD, nullptr, 0);
    gemm(stream, s, SD, Wv3, Hn * Pn * 3, v3, Hn * Pn * 3, Lc, Hn * Pn * 3, SD, nullptr, 0);

    k_points<<<cdiv(Lc * Hn, 256), 256, 0, stream>>>(q1, k1, q2, k2, v3, rot, trn,
                                                     gamma, qext, kext, qn, kn, v3t);

    k_att<<<cdiv((Lc / 16) * (Lc / 16) * Hn, 8), 256, 0, stream>>>(
        qext, kext, qn, kn, bz, gamma, att);

    k_o1<<<cdiv(Lc * (ZD / 16), 8), 256, 0, stream>>>(att, zN, o1);
    k_o2<<<cdiv((Lc / 16) * Hn, 8), 256, 0, stream>>>(att, v1, o2);
    k_tmp<<<cdiv((Lc / 16) * 2 * Hn, 8), 256, 0, stream>>>(att, v3t, tb);
    k_o3<<<cdiv(Lc * Hn, 256), 256, 0, stream>>>(tb, rot, trn, o3);

    // s += o1@Wl1 + o2@Wl2 + o3@Wl3 + biases  (residual: s already holds old s)
    gemm(stream, o1, Hn * ZD, Wl1, SD, s, SD, Lc, SD, Hn * ZD, bl1, 2);
    gemm(stream, o2, Hn * Cn, Wl2, SD, s, SD, Lc, SD, Hn * Cn, bl2, 2);
    gemm(stream, o3, Hn * Pn * 3, Wl3, SD, s, SD, Lc, SD, Hn * Pn * 3, bl3, 2);

    // transition
    k_ln_wave<<<cdiv(Lc, 8), 256, 0, stream>>>(s, t0, tn1g, tn1b, Lc, SD);
    gemm(stream, t0, SD, Wt1, SD, t1, SD, Lc, SD, SD, bt1, 1);
    gemm(stream, t1, SD, Wt2, SD, t2, SD, Lc, SD, SD, bt2, 1);
    gemm(stream, t2, SD, Wt3, SD, s, SD, Lc, SD, SD, bt3, 2);
    k_ln_wave<<<cdiv(Lc, 8), 256, 0, stream>>>(s, s, tn2g, tn2b, Lc, SD);

    // backbone update
    gemm(stream, s, SD, Wbb, 6, pred, 6, Lc, 6, SD, bbb, 0);
    k_frame_update<<<cdiv(Lc, 256), 256, 0, stream>>>(pred, rot, trn);
  }
}
